// DecoderBlock_48816598286524
// MI455X (gfx1250) — matrix operand order repaired and checked by reading, not compiled, not run on hardware
//
#include <hip/hip_runtime.h>
#include <math.h>

// Problem constants (match reference)
#define B_   2
#define S_   2048
#define D_   1024
#define H_   16
#define DH_  64
#define ROWS (B_ * S_)   // 4096

typedef __bf16 bf16;
typedef __attribute__((ext_vector_type(16))) __bf16 v16bf;
typedef __attribute__((ext_vector_type(8)))  __bf16 v8bf;
typedef __attribute__((ext_vector_type(8)))  float  v8f;

static __device__ __forceinline__ v16bf ld_frag(const bf16* lo, const bf16* hi) {
  v8bf a = *(const v8bf*)lo;
  v8bf b = *(const v8bf*)hi;
  return __builtin_shufflevector(a, b, 0,1,2,3,4,5,6,7,8,9,10,11,12,13,14,15);
}

static __device__ __forceinline__ float gelu_tanh_f(float x) {
  float x3 = x * x * x;
  return 0.5f * x * (1.0f + tanhf(0.7978845608028654f * (x + 0.044715f * x3)));
}

// Low 32 bits of a flat shared-space address == LDS byte offset (aperture is in addr[63:32]).
static __device__ __forceinline__ unsigned lds_off(const void* p) {
  return (unsigned)(uintptr_t)p;
}

// One 16B async Global->LDS copy per lane (GVS mode: SGPR64 base + VGPR32 byte offset).
static __device__ __forceinline__ void async_copy16(unsigned lds, const bf16* base, unsigned goff) {
  asm volatile("global_load_async_to_lds_b128 %0, %1, %2"
               :: "v"(lds), "v"(goff), "s"(base)
               : "memory");
}

// ---------------------------------------------------------------------------
// Elementwise fp32 -> bf16 convert
// ---------------------------------------------------------------------------
__global__ void convert_bf16_kernel(const float* __restrict__ in, bf16* __restrict__ out, size_t n) {
  size_t i = (size_t)blockIdx.x * blockDim.x + threadIdx.x;
  if (i < n) out[i] = (bf16)in[i];
}

// in: [K, N] fp32 row-major -> out: [N, K] bf16 row-major
__global__ void transpose_bf16_kernel(const float* __restrict__ in, bf16* __restrict__ out, int K, int N) {
  size_t i = (size_t)blockIdx.x * blockDim.x + threadIdx.x;
  size_t total = (size_t)K * N;
  if (i >= total) return;
  int n = (int)(i / K);
  int k = (int)(i % K);
  out[i] = (bf16)in[(size_t)k * N + n];
}

// ---------------------------------------------------------------------------
// WMMA GEMM with async Global->LDS double-buffered staging.
// C[M,N] = A[M,K](bf16) * Bt[N,K]^T(bf16) + bias
// EPI 0: store fp32; EPI 1: gelu then store bf16
// Block: 8 waves -> 128(M) x 256(N); each wave owns a 64x64 tile.
// LDS rows padded to 40 bf16 (80B) for conflict-free 16B fragment reads.
// ---------------------------------------------------------------------------
#define LDSPITCH 40

template <int EPI>
__global__ __launch_bounds__(256) void gemm_bf16_wmma(
    const bf16* __restrict__ A, const bf16* __restrict__ Bt,
    const float* __restrict__ bias,
    float* __restrict__ Cf, bf16* __restrict__ Cb,
    int M, int N, int K) {
  __shared__ bf16 Alds[2][128][LDSPITCH];
  __shared__ bf16 Blds[2][256][LDSPITCH];

  const int t    = threadIdx.x;
  const int lane = t & 31;
  const int wave = t >> 5;
  const int hlf  = lane >> 4;   // 16-lane half of the wave
  const int n16  = lane & 15;
  const int tileM = blockIdx.x * 128 + (wave & 1) * 64;
  const int tileN = blockIdx.y * 256 + (wave >> 1) * 64;

  // --- async-copy work assignment: 16B chunks; A tile 128x32 (2/thread), B tile 256x32 (4/thread)
  unsigned gA[2], lA[2];
#pragma unroll
  for (int u = 0; u < 2; ++u) {
    int c = t + 256 * u, row = c >> 2, part = c & 3;
    gA[u] = (unsigned)((blockIdx.x * 128 + row) * K + part * 8) * 2u;  // byte offset into A
    lA[u] = (unsigned)(row * LDSPITCH + part * 8) * 2u;                // byte offset into tile
  }
  unsigned gB[4], lB[4];
#pragma unroll
  for (int u = 0; u < 4; ++u) {
    int c = t + 256 * u, row = c >> 2, part = c & 3;
    gB[u] = (unsigned)((blockIdx.y * 256 + row) * K + part * 8) * 2u;
    lB[u] = (unsigned)(row * LDSPITCH + part * 8) * 2u;
  }
  const unsigned ldsAbase[2] = { lds_off(&Alds[0][0][0]), lds_off(&Alds[1][0][0]) };
  const unsigned ldsBbase[2] = { lds_off(&Blds[0][0][0]), lds_off(&Blds[1][0][0]) };

  auto issue = [&](int buf, int ktElems) {
    unsigned kb2 = (unsigned)ktElems * 2u;
#pragma unroll
    for (int u = 0; u < 2; ++u) async_copy16(ldsAbase[buf] + lA[u], A,  gA[u] + kb2);
#pragma unroll
    for (int u = 0; u < 4; ++u) async_copy16(ldsBbase[buf] + lB[u], Bt, gB[u] + kb2);
  };

  v8f acc[4][4] = {};
  const int nk = K / 32;

  issue(0, 0);
  issue(1, 32);

  for (int ki = 0; ki < nk; ++ki) {
    // current buffer landed (6 per-wave async ops may remain in flight for the next buffer)
    if (ki + 1 < nk) asm volatile("s_wait_asynccnt 0x6" ::: "memory");
    else             asm volatile("s_wait_asynccnt 0x0" ::: "memory");
    __syncthreads();

    const int buf = ki & 1;
    v16bf af[4], bfr[4];
#pragma unroll
    for (int i = 0; i < 4; ++i) {
      // A frag (16x32 interleaved): elems 0..7 -> K=8h+i, 8..15 -> K=16+8h+i
      const bf16* p = &Alds[buf][(wave & 1) * 64 + i * 16 + n16][8 * hlf];
      af[i] = ld_frag(p, p + 16);
    }
#pragma unroll
    for (int j = 0; j < 4; ++j) {
      // B frag (32x16 interleaved): elems 0..7 -> K=8h+i, 8..15 -> K=16+8h+i
      const bf16* p = &Blds[buf][(wave >> 1) * 64 + j * 16 + n16][8 * hlf];
      bfr[j] = ld_frag(p, p + 16);
    }
#pragma unroll
    for (int i = 0; i < 4; ++i)
#pragma unroll
      for (int j = 0; j < 4; ++j)
        acc[i][j] = __builtin_amdgcn_wmma_f32_16x16x32_bf16(
            false, af[i], false, bfr[j], (short)0, acc[i][j], false, false);

    __syncthreads();  // everyone done reading buf before it is overwritten
    if (ki + 2 < nk) issue(buf, (ki + 2) * 32);
  }

  // Epilogue. C/D layout: elem e -> row = 8*hlf + e, col = n16 (within 16x16 tile)
#pragma unroll
  for (int j = 0; j < 4; ++j) {
    const int col = tileN + j * 16 + n16;
    const float bv = bias[col];
#pragma unroll
    for (int i = 0; i < 4; ++i) {
#pragma unroll
      for (int e = 0; e < 8; ++e) {
        const int row = tileM + i * 16 + 8 * hlf + e;
        float v = acc[i][j][e] + bv;
        if (EPI == 1) {
          Cb[(size_t)row * N + col] = (bf16)gelu_tanh_f(v);
        } else {
          Cf[(size_t)row * N + col] = v;
        }
      }
    }
  }
}

// ---------------------------------------------------------------------------
// Split qkv fp32 [ROWS, 3D] into q/k bf16 [B,H,S,DH] and v bf16 [B,H,DH,S]
// ---------------------------------------------------------------------------
__global__ void qkv_split_kernel(const float* __restrict__ qkv,
                                 bf16* __restrict__ qb, bf16* __restrict__ kb, bf16* __restrict__ vb) {
  size_t idx = (size_t)blockIdx.x * blockDim.x + threadIdx.x;
  if (idx >= (size_t)ROWS * D_) return;
  int r  = (int)(idx / D_);     // b*S + s
  int d0 = (int)(idx % D_);
  int b  = r / S_;
  int s  = r % S_;
  int hh = d0 >> 6;
  int d  = d0 & 63;
  const float* base = qkv + (size_t)r * (3 * D_);
  size_t qkoff = ((size_t)(b * H_ + hh) * S_ + s) * DH_ + d;
  qb[qkoff] = (bf16)base[d0];
  kb[qkoff] = (bf16)base[D_ + d0];
  vb[((size_t)(b * H_ + hh) * DH_ + d) * S_ + s] = (bf16)base[2 * D_ + d0];
}

// ---------------------------------------------------------------------------
// Flash attention: 1 wave per (b,h, 16-row q tile); 32-key chunks.
// q,k: [B,H,S,DH] bf16; v: [B,H,DH,S] bf16; z out: [ROWS, D] bf16
// ---------------------------------------------------------------------------
__global__ __launch_bounds__(32) void attn_flash_kernel(
    const bf16* __restrict__ qb, const bf16* __restrict__ kb,
    const bf16* __restrict__ vb, bf16* __restrict__ zb) {
  __shared__ float plds[16][33];

  const int qt  = blockIdx.x;        // q tile (16 rows)
  const int bh  = blockIdx.y;        // b*H + h
  const int b   = bh >> 4;
  const int hh  = bh & 15;
  const int lane = threadIdx.x;
  const int hlf  = lane >> 4;
  const int n16  = lane & 15;

  // Q fragments (A layout), rows qt*16..+15, K-dim = head dim 64 -> 2 frags
  const bf16* qbase = qb + ((size_t)bh * S_ + qt * 16 + n16) * DH_ + 8 * hlf;
  v16bf qf[2];
#pragma unroll
  for (int kk = 0; kk < 2; ++kk) qf[kk] = ld_frag(qbase + 32 * kk, qbase + 32 * kk + 16);

  v8f oacc[4] = {};
  float mrow[8], lrow[8];
#pragma unroll
  for (int e = 0; e < 8; ++e) { mrow[e] = -__builtin_inff(); lrow[e] = 0.0f; }

  const int qend = qt * 16 + 15;
  for (int c = 0; c <= qend; c += 32) {
    // K fragments (B layout): column = key index, K-dim = head dim (interleaved like A)
    const bf16* kbase = kb + ((size_t)bh * S_ + c + n16) * DH_ + 8 * hlf;
    v8f s0 = {}, s1 = {};
#pragma unroll
    for (int kk = 0; kk < 2; ++kk) {
      v16bf k0 = ld_frag(kbase + 32 * kk,            kbase + 32 * kk + 16);
      v16bf k1 = ld_frag(kbase + 16 * DH_ + 32 * kk, kbase + 16 * DH_ + 32 * kk + 16);
      s0 = __builtin_amdgcn_wmma_f32_16x16x32_bf16(false, qf[kk], false, k0, (short)0, s0, false, false);
      s1 = __builtin_amdgcn_wmma_f32_16x16x32_bf16(false, qf[kk], false, k1, (short)0, s1, false, false);
    }

    // scale + causal mask + online softmax update
#pragma unroll
    for (int e = 0; e < 8; ++e) {
      const int row = qt * 16 + 8 * hlf + e;
      float v0 = s0[e] * 0.125f;                 // 1/sqrt(64)
      float v1 = s1[e] * 0.125f;
      if (c + n16      > row) v0 = -__builtin_inff();
      if (c + 16 + n16 > row) v1 = -__builtin_inff();
      float rmax = fmaxf(v0, v1);
#pragma unroll
      for (int off = 8; off > 0; off >>= 1) rmax = fmaxf(rmax, __shfl_xor(rmax, off, 32));
      float mnew  = fmaxf(mrow[e], rmax);
      float alpha = __expf(mrow[e] - mnew);
      float p0 = __expf(v0 - mnew);
      float p1 = __expf(v1 - mnew);
      float rsum = p0 + p1;
#pragma unroll
      for (int off = 8; off > 0; off >>= 1) rsum += __shfl_xor(rsum, off, 32);
      lrow[e] = lrow[e] * alpha + rsum;
      mrow[e] = mnew;
#pragma unroll
      for (int j = 0; j < 4; ++j) oacc[j][e] *= alpha;
      plds[8 * hlf + e][n16]      = p0;
      plds[8 * hlf + e][16 + n16] = p1;
    }
    __syncthreads();

    // Rebuild P as an A-layout bf16 fragment (row m = n16)
    v16bf pf;
    const float* prow = &plds[n16][0];
#pragma unroll
    for (int i = 0; i < 8; ++i) {
      pf[i]     = (bf16)prow[8 * hlf + i];
      pf[8 + i] = (bf16)prow[16 + 8 * hlf + i];
    }
    __syncthreads();

    // V fragments (B layout): column = head dim, K-dim = key index (interleaved like A, via [b,h,d,s])
    const bf16* vbase = vb + ((size_t)bh * DH_ + n16) * S_ + c + 8 * hlf;
#pragma unroll
    for (int j = 0; j < 4; ++j) {
      const bf16* vp = vbase + (size_t)j * 16 * S_;
      v16bf vf = ld_frag(vp, vp + 16);
      oacc[j] = __builtin_amdgcn_wmma_f32_16x16x32_bf16(false, pf, false, vf, (short)0, oacc[j], false, false);
    }
  }

  // Normalize and store z
#pragma unroll
  for (int j = 0; j < 4; ++j) {
    const int col = hh * DH_ + j * 16 + n16;
#pragma unroll
    for (int e = 0; e < 8; ++e) {
      const int row = qt * 16 + 8 * hlf + e;
      float inv = 1.0f / lrow[e];
      zb[(size_t)(b * S_ + row) * D_ + col] = (bf16)(oacc[j][e] * inv);
    }
  }
}

// ---------------------------------------------------------------------------
// y = LayerNorm(xin)*gamma + beta + resid ; row length = 1024, 1 block/row
// ---------------------------------------------------------------------------
__global__ __launch_bounds__(256) void ln_residual_kernel(
    const float* __restrict__ xin, const float* __restrict__ gamma, const float* __restrict__ beta,
    const float* __restrict__ resid, float* __restrict__ outf, bf16* __restrict__ outb) {
  __shared__ float sh1[256], sh2[256];
  const int row = blockIdx.x;
  const int t = threadIdx.x;
  const float* xr = xin + (size_t)row * D_;
  float v[4], s = 0.0f, sq = 0.0f;
#pragma unroll
  for (int k = 0; k < 4; ++k) {
    v[k] = xr[t + 256 * k];
    s += v[k];
    sq += v[k] * v[k];
  }
  sh1[t] = s; sh2[t] = sq;
  __syncthreads();
  for (int off = 128; off > 0; off >>= 1) {
    if (t < off) { sh1[t] += sh1[t + off]; sh2[t] += sh2[t + off]; }
    __syncthreads();
  }
  const float mean = sh1[0] * (1.0f / D_);
  const float var  = sh2[0] * (1.0f / D_) - mean * mean;
  const float rstd = rsqrtf(var + 1e-5f);
#pragma unroll
  for (int k = 0; k < 4; ++k) {
    const int col = t + 256 * k;
    float y = (v[k] - mean) * rstd * gamma[col] + beta[col] + resid[(size_t)row * D_ + col];
    outf[(size_t)row * D_ + col] = y;
    if (outb) outb[(size_t)row * D_ + col] = (bf16)y;
  }
}

// ---------------------------------------------------------------------------
extern "C" void kernel_launch(void* const* d_in, const int* in_sizes, int n_in,
                              void* d_out, int out_size, void* d_ws, size_t ws_size,
                              hipStream_t stream) {
  (void)in_sizes; (void)n_in; (void)out_size; (void)ws_size;
  const float* x    = (const float*)d_in[0];
  const float* Wqkv = (const float*)d_in[1];
  const float* bqkv = (const float*)d_in[2];
  const float* Wo   = (const float*)d_in[3];
  const float* bo   = (const float*)d_in[4];
  const float* W1   = (const float*)d_in[5];
  const float* b1   = (const float*)d_in[6];
  const float* W2   = (const float*)d_in[7];
  const float* b2   = (const float*)d_in[8];
  const float* ln1w = (const float*)d_in[9];
  const float* ln1b = (const float*)d_in[10];
  const float* ln2w = (const float*)d_in[11];
  const float* ln2b = (const float*)d_in[12];
  float* out = (float*)d_out;

  char* ws = (char*)d_ws;
  size_t o = 0;
  auto take = [&](size_t n) { size_t r = o; o += (n + 255) & ~(size_t)255; return r; };

  bf16*  xb    = (bf16*)(ws + take((size_t)ROWS * D_ * 2));          // 8 MiB
  bf16*  WqkvT = (bf16*)(ws + take((size_t)3 * D_ * D_ * 2));        // 6 MiB
  bf16*  WoT   = (bf16*)(ws + take((size_t)D_ * D_ * 2));            // 2 MiB
  bf16*  W1T   = (bf16*)(ws + take((size_t)4 * D_ * D_ * 2));        // 8 MiB
  bf16*  W2T   = (bf16*)(ws + take((size_t)4 * D_ * D_ * 2));        // 8 MiB
  size_t qkv_off = take((size_t)ROWS * 3 * D_ * 4);                  // 48 MiB
  float* qkv32 = (float*)(ws + qkv_off);
  bf16*  hb    = (bf16*)(ws + qkv_off);                              // alias: MLP hidden (32 MiB)
  float* mlp32 = (float*)(ws + qkv_off + (size_t)ROWS * 4 * D_ * 2); // alias: +32 MiB (16 MiB)
  size_t qb_off = take((size_t)B_ * H_ * S_ * DH_ * 2);              // 8 MiB
  bf16*  qb    = (bf16*)(ws + qb_off);
  bf16*  kb    = (bf16*)(ws + take((size_t)B_ * H_ * S_ * DH_ * 2)); // 8 MiB (contiguous after qb)
  size_t vb_off = take((size_t)B_ * H_ * S_ * DH_ * 2);              // 8 MiB
  bf16*  vb    = (bf16*)(ws + vb_off);
  bf16*  zb    = (bf16*)(ws + take((size_t)ROWS * D_ * 2));          // 8 MiB
  float* out1f = (float*)(ws + take((size_t)ROWS * D_ * 4));         // 16 MiB
  float* attn32 = (float*)(ws + qb_off);                             // alias q+k after attention
  bf16*  out1b  = (bf16*)(ws + vb_off);                              // alias v after attention

  const int T = 256;
  // 1) convert x + transpose-convert weights to bf16 [N,K]
  convert_bf16_kernel<<<(ROWS * D_ + T - 1) / T, T, 0, stream>>>(x, xb, (size_t)ROWS * D_);
  transpose_bf16_kernel<<<(3 * D_ * D_ + T - 1) / T, T, 0, stream>>>(Wqkv, WqkvT, D_, 3 * D_);
  transpose_bf16_kernel<<<(D_ * D_ + T - 1) / T, T, 0, stream>>>(Wo, WoT, D_, D_);
  transpose_bf16_kernel<<<(4 * D_ * D_ + T - 1) / T, T, 0, stream>>>(W1, W1T, D_, 4 * D_);
  transpose_bf16_kernel<<<(4 * D_ * D_ + T - 1) / T, T, 0, stream>>>(W2, W2T, 4 * D_, D_);

  // 2) qkv = x @ Wqkv + bqkv
  gemm_bf16_wmma<0><<<dim3(ROWS / 128, (3 * D_) / 256), T, 0, stream>>>(
      xb, WqkvT, bqkv, qkv32, nullptr, ROWS, 3 * D_, D_);

  // 3) split/convert to per-head layouts
  qkv_split_kernel<<<(ROWS * D_ + T - 1) / T, T, 0, stream>>>(qkv32, qb, kb, vb);

  // 4) flash attention -> z
  attn_flash_kernel<<<dim3(S_ / 16, B_ * H_), 32, 0, stream>>>(qb, kb, vb, zb);

  // 5) attn_out = z @ Wo + bo
  gemm_bf16_wmma<0><<<dim3(ROWS / 128, D_ / 256), T, 0, stream>>>(
      zb, WoT, bo, attn32, nullptr, ROWS, D_, D_);

  // 6) out1 = LN(attn_out)*w+b + x
  ln_residual_kernel<<<ROWS, T, 0, stream>>>(attn32, ln1w, ln1b, x, out1f, out1b);

  // 7) h = gelu(out1 @ W1 + b1)  (bf16 out)
  gemm_bf16_wmma<1><<<dim3(ROWS / 128, (4 * D_) / 256), T, 0, stream>>>(
      out1b, W1T, b1, nullptr, hb, ROWS, 4 * D_, D_);

  // 8) mlp = h @ W2 + b2
  gemm_bf16_wmma<0><<<dim3(ROWS / 128, D_ / 256), T, 0, stream>>>(
      hb, W2T, b2, mlp32, nullptr, ROWS, D_, 4 * D_);

  // 9) out = LN(mlp)*w+b + out1
  ln_residual_kernel<<<ROWS, T, 0, stream>>>(mlp32, ln2w, ln2b, out1f, out, nullptr);
}
